// SimpleModelQ_3401614098521
// MI455X (gfx1250) — matrix operand order repaired and checked by reading, not compiled, not run on hardware
//
#include <hip/hip_runtime.h>
#include <hip/hip_bf16.h>

typedef __attribute__((ext_vector_type(16))) _Float16 v16h;
typedef __attribute__((ext_vector_type(8)))  float    v8f;

#define FEATN  255
#define LASTN  258
#define HID    1024
#define BATCH  128
#define SEQ    1024

// ---------------- workspace layout (bytes); total ~3.1 MB ----------------
static const size_t OFF_VW16 = 0;                      // 1024*256 f16 = 512 KB
static const size_t OFF_W116 = 524288;                 // 512*1024 f16 = 1 MB
static const size_t OFF_W216 = 1572864;                // 256*512  f16 = 256 KB
static const size_t OFF_W316 = 1835008;                // 16*256   f16 = 8 KB (rows>=2 zero)
static const size_t OFF_QK   = 1843200;                // 128*256  f32
static const size_t OFF_QKB  = 1974272;                // 128      f32
static const size_t OFF_SC   = 1974784;                // 128*1024 f32
static const size_t OFF_AT   = 2499072;                // 128*1024 f32
static const size_t OFF_FB   = 3023360;                // 128*256  f16

// ---------------- weight convert + zero-pad to f16 ----------------
__global__ __launch_bounds__(256) void k_cvt_pad(const float* __restrict__ src,
                                                 _Float16* __restrict__ dst,
                                                 int src_rows, int src_cols,
                                                 int dst_cols, int total) {
  int idx = blockIdx.x * 256 + threadIdx.x;
  if (idx >= total) return;
  int r = idx / dst_cols, c = idx % dst_cols;
  float v = (r < src_rows && c < src_cols) ? src[r * src_cols + c] : 0.f;
  dst[idx] = (_Float16)v;
}

// ---------------- q = archi@qW^T + qb ; qk[b,f] = sum_h q[h]*kW[h,f] ----------------
__global__ __launch_bounds__(256) void k_q_qk(const float* __restrict__ in,
                                              const float* __restrict__ kW,
                                              const float* __restrict__ kb,
                                              const float* __restrict__ qW,
                                              const float* __restrict__ qb,
                                              float* __restrict__ qk,
                                              float* __restrict__ qkb) {
  __shared__ float qv[HID];
  __shared__ float red[256];
  int b = blockIdx.x, t = threadIdx.x;
  const float* row0 = in + (size_t)b * SEQ * LASTN;
  float a0 = row0[FEATN], a1 = row0[FEATN + 1], a2 = row0[FEATN + 2];
  float pk = 0.f;
#pragma unroll
  for (int i = 0; i < 4; ++i) {
    int h = t + 256 * i;
    float qh = qb[h] + a0 * qW[h * 3 + 0] + a1 * qW[h * 3 + 1] + a2 * qW[h * 3 + 2];
    qv[h] = qh;
    pk += qh * kb[h];
  }
  red[t] = pk;
  __syncthreads();
  for (int s2 = 128; s2 > 0; s2 >>= 1) {
    if (t < s2) red[t] += red[t + s2];
    __syncthreads();
  }
  const float scale = rsqrtf((float)LASTN);   // d_k = 258
  if (t == 0) qkb[b] = red[0] * scale;
  if (t < FEATN) {
    float s = 0.f;
    for (int h = 0; h < HID; ++h) s += qv[h] * kW[h * FEATN + t];  // coalesced in t
    qk[b * 256 + t] = s * scale;
  } else if (t == FEATN) {
    qk[b * 256 + t] = 0.f;
  }
}

// ---------------- scores[b,s] = feats[b,s,:]·qk[b,:] + qkb[b]  (pass 1) ----------------
__global__ __launch_bounds__(256) void k_scores(const float* __restrict__ in,
                                                const float* __restrict__ qk,
                                                const float* __restrict__ qkb,
                                                float* __restrict__ scores) {
  __shared__ float qkl[256];
  int b = blockIdx.x, t = threadIdx.x;
  qkl[t] = qk[b * 256 + t];
  __syncthreads();
  int w = t >> 5, lane = t & 31;
  float cb = qkb[b];
  for (int s = w; s < SEQ; s += 8) {
    const float* row = in + ((size_t)b * SEQ + s) * LASTN;
    float acc = 0.f;
#pragma unroll
    for (int i = 0; i < 8; ++i) {
      int f = lane + 32 * i;
      if (f < FEATN) acc += qkl[f] * row[f];
    }
#pragma unroll
    for (int off = 16; off > 0; off >>= 1) acc += __shfl_xor(acc, off, 32);
    if (lane == 0) scores[(size_t)b * SEQ + s] = acc + cb;
  }
}

// ---------------- softmax over S per batch ----------------
__global__ __launch_bounds__(256) void k_softmax(const float* __restrict__ scores,
                                                 float* __restrict__ attn) {
  __shared__ float red[256];
  int b = blockIdx.x, t = threadIdx.x;
  float v[4];
  float mx = -3.4e38f;
#pragma unroll
  for (int i = 0; i < 4; ++i) {
    v[i] = scores[(size_t)b * SEQ + t + 256 * i];
    mx = fmaxf(mx, v[i]);
  }
  red[t] = mx;
  __syncthreads();
  for (int s2 = 128; s2 > 0; s2 >>= 1) {
    if (t < s2) red[t] = fmaxf(red[t], red[t + s2]);
    __syncthreads();
  }
  mx = red[0];
  __syncthreads();
  float sum = 0.f;
#pragma unroll
  for (int i = 0; i < 4; ++i) {
    v[i] = __expf(v[i] - mx);
    sum += v[i];
  }
  red[t] = sum;
  __syncthreads();
  for (int s2 = 128; s2 > 0; s2 >>= 1) {
    if (t < s2) red[t] += red[t + s2];
    __syncthreads();
  }
  float inv = 1.f / red[0];
#pragma unroll
  for (int i = 0; i < 4; ++i) attn[(size_t)b * SEQ + t + 256 * i] = v[i] * inv;
}

// ---------------- fbar[b,f] = sum_s attn[b,s]*feats[b,s,f]  (pass 2, f16 out) ----------------
__global__ __launch_bounds__(256) void k_fbar(const float* __restrict__ in,
                                              const float* __restrict__ attn,
                                              _Float16* __restrict__ fbar16) {
  __shared__ float at[SEQ];
  int b = blockIdx.x, t = threadIdx.x;
#pragma unroll
  for (int i = 0; i < 4; ++i) at[t + 256 * i] = attn[(size_t)b * SEQ + t + 256 * i];
  __syncthreads();
  const float* base = in + (size_t)b * SEQ * LASTN;
  float a0 = 0, a1 = 0, a2 = 0, a3 = 0;
  if (t < FEATN) {
    for (int s = 0; s < SEQ; s += 4) {
      a0 += at[s + 0] * base[(size_t)(s + 0) * LASTN + t];
      a1 += at[s + 1] * base[(size_t)(s + 1) * LASTN + t];
      a2 += at[s + 2] * base[(size_t)(s + 2) * LASTN + t];
      a3 += at[s + 3] * base[(size_t)(s + 3) * LASTN + t];
    }
  }
  float acc = (a0 + a1) + (a2 + a3);
  fbar16[b * 256 + t] = (t < FEATN) ? (_Float16)acc : (_Float16)0.f;
}

// ---------------- WMMA fragment helpers (contiguous-k layouts, wave32) ----------------
// A 16x32 f16: lane row M=lane%16; element e -> K = e + 16*(lane/16)
//   => dword pair j: K = 2j + 16*half : two contiguous f16
__device__ inline v16h load_a_frag(const _Float16* base, int ldk, int row0, int k0, int lane) {
  int m = row0 + (lane & 15);
  int half = lane >> 4;
  const unsigned int* p = (const unsigned int*)(base + (size_t)m * ldk + k0 + 16 * half);
  const unsigned int* q = (const unsigned int*)(base + (size_t)m * ldk + k0 + 8 + 16 * half);
  v16h a;
  unsigned int* au = (unsigned int*)&a;
#pragma unroll
  for (int j = 0; j < 4; ++j) au[j] = p[j];
#pragma unroll
  for (int j = 0; j < 4; ++j) au[4 + j] = q[j];
  return a;
}
// B 32x16 f16: lane col N=lane%16; element e -> K = e + 16*(lane/16): 8 contiguous dwords
__device__ inline v16h load_b_frag(const _Float16* base, int ldk, int n0, int k0, int lane) {
  int n = n0 + (lane & 15);
  int half = lane >> 4;
  const unsigned int* p = (const unsigned int*)(base + (size_t)n * ldk + k0 + 16 * half);
  v16h bb;
  unsigned int* bu = (unsigned int*)&bb;
#pragma unroll
  for (int j = 0; j < 8; ++j) bu[j] = p[j];
  return bb;
}
__device__ inline v8f wmma16(v16h a, v16h b, v8f c) {
  return __builtin_amdgcn_wmma_f32_16x16x32_f16(false, a, false, b, (short)0, c, false, false);
}

// ---------------- fused: res = fbar@vW^T+vb ; 3-layer MLP ; out 128x2 ----------------
__global__ __launch_bounds__(256) void k_mlp(const _Float16* __restrict__ fbar16,
                                             const _Float16* __restrict__ vW16,
                                             const float* __restrict__ vb,
                                             const _Float16* __restrict__ W116,
                                             const float* __restrict__ b1,
                                             const _Float16* __restrict__ W216,
                                             const float* __restrict__ b2,
                                             const _Float16* __restrict__ W316,
                                             const float* __restrict__ b3,
                                             float* __restrict__ out) {
  __shared__ _Float16 s_res[16 * 1024];  // 32 KB
  __shared__ _Float16 s_h1[16 * 512];    // 16 KB
  __shared__ _Float16 s_h2[16 * 256];    //  8 KB
  int w = threadIdx.x >> 5, lane = threadIdx.x & 31;
  int half = lane >> 4, nl = lane & 15;
  int m0 = blockIdx.x * 16;

  // Stage A: res (M=16, N=1024, K=256); A frags held in registers (8 * v16h)
  v16h afr[8];
#pragma unroll
  for (int c = 0; c < 8; ++c) afr[c] = load_a_frag(fbar16, 256, m0, c * 32, lane);
  for (int t = w; t < 64; t += 8) {
    int n0 = t * 16;
    v8f acc = {};
#pragma unroll
    for (int c = 0; c < 8; ++c) {
      v16h bf = load_b_frag(vW16, 256, n0, c * 32, lane);
      acc = wmma16(afr[c], bf, acc);
    }
    int n = n0 + nl;
    float bias = vb[n];
#pragma unroll
    for (int r = 0; r < 8; ++r)
      s_res[(r + 8 * half) * 1024 + n] = (_Float16)(acc[r] + bias);
  }
  __syncthreads();

  // Stage B: h1 = relu(res@W1^T + b1)  (N=512, K=1024)
  for (int t = w; t < 32; t += 8) {
    int n0 = t * 16;
    v8f acc = {};
    for (int c = 0; c < 32; ++c) {
      v16h af = load_a_frag(s_res, 1024, 0, c * 32, lane);
      v16h bf = load_b_frag(W116, 1024, n0, c * 32, lane);
      acc = wmma16(af, bf, acc);
    }
    int n = n0 + nl;
    float bias = b1[n];
#pragma unroll
    for (int r = 0; r < 8; ++r) {
      float vv = acc[r] + bias;
      s_h1[(r + 8 * half) * 512 + n] = (_Float16)(vv > 0.f ? vv : 0.f);
    }
  }
  __syncthreads();

  // Stage C: h2 = relu(h1@W2^T + b2)  (N=256, K=512)
  for (int t = w; t < 16; t += 8) {
    int n0 = t * 16;
    v8f acc = {};
    for (int c = 0; c < 16; ++c) {
      v16h af = load_a_frag(s_h1, 512, 0, c * 32, lane);
      v16h bf = load_b_frag(W216, 512, n0, c * 32, lane);
      acc = wmma16(af, bf, acc);
    }
    int n = n0 + nl;
    float bias = b2[n];
#pragma unroll
    for (int r = 0; r < 8; ++r) {
      float vv = acc[r] + bias;
      s_h2[(r + 8 * half) * 256 + n] = (_Float16)(vv > 0.f ? vv : 0.f);
    }
  }
  __syncthreads();

  // Stage D: out = h2@W3^T + b3  (N=2 padded to 16, K=256); wave 0 only (EXEC all-1s in-wave)
  if (w == 0) {
    v8f acc = {};
#pragma unroll
    for (int c = 0; c < 8; ++c) {
      v16h af = load_a_frag(s_h2, 256, 0, c * 32, lane);
      v16h bf = load_b_frag(W316, 256, 0, c * 32, lane);
      acc = wmma16(af, bf, acc);
    }
    if (nl < 2) {
      float bias = b3[nl];
#pragma unroll
      for (int r = 0; r < 8; ++r)
        out[(m0 + r + 8 * half) * 2 + nl] = acc[r] + bias;
    }
  }
}

extern "C" void kernel_launch(void* const* d_in, const int* in_sizes, int n_in,
                              void* d_out, int out_size, void* d_ws, size_t ws_size,
                              hipStream_t stream) {
  (void)in_sizes; (void)n_in; (void)out_size; (void)ws_size;  // needs ~3.1 MB ws
  const float* in = (const float*)d_in[0];
  const float* kW = (const float*)d_in[1];
  const float* kb = (const float*)d_in[2];
  const float* vW = (const float*)d_in[3];
  const float* vb = (const float*)d_in[4];
  const float* qW = (const float*)d_in[5];
  const float* qb = (const float*)d_in[6];
  const float* W1 = (const float*)d_in[7];
  const float* b1 = (const float*)d_in[8];
  const float* W2 = (const float*)d_in[9];
  const float* b2 = (const float*)d_in[10];
  const float* W3 = (const float*)d_in[11];
  const float* b3 = (const float*)d_in[12];
  float* out = (float*)d_out;
  char* ws = (char*)d_ws;

  _Float16* vW16 = (_Float16*)(ws + OFF_VW16);
  _Float16* W116 = (_Float16*)(ws + OFF_W116);
  _Float16* W216 = (_Float16*)(ws + OFF_W216);
  _Float16* W316 = (_Float16*)(ws + OFF_W316);
  float* qk = (float*)(ws + OFF_QK);
  float* qkb = (float*)(ws + OFF_QKB);
  float* scores = (float*)(ws + OFF_SC);
  float* attn = (float*)(ws + OFF_AT);
  _Float16* fbar16 = (_Float16*)(ws + OFF_FB);

  k_cvt_pad<<<(1024 * 256 + 255) / 256, 256, 0, stream>>>(vW, vW16, 1024, 255, 256, 1024 * 256);
  k_cvt_pad<<<(512 * 1024 + 255) / 256, 256, 0, stream>>>(W1, W116, 512, 1024, 1024, 512 * 1024);
  k_cvt_pad<<<(256 * 512 + 255) / 256, 256, 0, stream>>>(W2, W216, 256, 512, 512, 256 * 512);
  k_cvt_pad<<<(16 * 256 + 255) / 256, 256, 0, stream>>>(W3, W316, 2, 256, 256, 16 * 256);

  k_q_qk<<<BATCH, 256, 0, stream>>>(in, kW, kb, qW, qb, qk, qkb);
  k_scores<<<BATCH, 256, 0, stream>>>(in, qk, qkb, scores);
  k_softmax<<<BATCH, 256, 0, stream>>>(scores, attn);
  k_fbar<<<BATCH, 256, 0, stream>>>(in, attn, fbar16);
  k_mlp<<<8, 256, 0, stream>>>(fbar16, vW16, vb, W116, b1, W216, b2, W316, b3, out);
}
